// MNIST_GAT3_10668698763721
// MI455X (gfx1250) — matrix operand order repaired and checked by reading, not compiled, not run on hardware
//
#include <hip/hip_runtime.h>
#include <hip/hip_bf16.h>
#include <math.h>

// ============================================================================
// MI455X (gfx1250) implementation of GATv2 x4 + SetTransformer + MLP head.
// GEMMs run through v_wmma_f32_16x16x32_bf16 (wave32 WMMA) with register
// double-buffering; edge softmax / aggregation through vectorized VALU
// kernels with f32 atomics.
// ============================================================================

typedef __bf16 bf16_t;
typedef bf16_t v16bf __attribute__((ext_vector_type(16)));
typedef float  v8f   __attribute__((ext_vector_type(8)));

#define D_NTOT   64000
#define D_E      256000
#define D_ETOT   (D_E + D_NTOT)
#define D_NB     640
#define D_NPER   100

// ---------------- workspace layout (bytes) ----------------
static const size_t SZ_BIG    = (size_t)D_NTOT * 384 * 4;           // 98.3 MB
static const size_t OFF_HA    = 0;
static const size_t OFF_HB    = OFF_HA + SZ_BIG;
static const size_t OFF_XL    = OFF_HB + SZ_BIG;
static const size_t OFF_XR    = OFF_XL + SZ_BIG;
static const size_t OFF_ABF   = OFF_XR + SZ_BIG;                    // bf16 A scratch
static const size_t OFF_WT    = OFF_ABF + (size_t)D_NTOT * 384 * 2; // bf16 Wt scratch
static const size_t OFF_ALPHA = OFF_WT + (size_t)384 * 384 * 2;
static const size_t OFF_AMAX  = OFF_ALPHA + (size_t)D_ETOT * 3 * 4;
static const size_t OFF_DENOM = OFF_AMAX + (size_t)D_NTOT * 3 * 4;
static const size_t OFF_QPMA  = OFF_DENOM + (size_t)D_NTOT * 3 * 4;
static const size_t OFF_TA    = OFF_QPMA + 256;
static const size_t OFF_TB    = OFF_TA + (size_t)D_NB * 64 * 4;
static const size_t OFF_TC    = OFF_TB + (size_t)D_NB * 64 * 4;
// STA sub-buffers carved out of the (then-free) XL region:
static const size_t OFF_Q     = OFF_XL;
static const size_t OFF_K     = OFF_XL + 1 * (size_t)D_NTOT * 64 * 4;
static const size_t OFF_V     = OFF_XL + 2 * (size_t)D_NTOT * 64 * 4;
static const size_t OFF_AO    = OFF_XL + 3 * (size_t)D_NTOT * 64 * 4;

// ---------------- device helpers ----------------
__device__ __forceinline__ unsigned short f2bf(float f) {
  unsigned u = __float_as_uint(f);
  u += 0x7FFFu + ((u >> 16) & 1u);            // round-to-nearest-even
  return (unsigned short)(u >> 16);
}

__device__ __forceinline__ void atomicMaxF(float* address, float val) {
  unsigned int* ai = (unsigned int*)address;
  unsigned int old = *ai;
  while (__uint_as_float(old) < val) {
    unsigned int assumed = old;
    old = atomicCAS(ai, assumed, __float_as_uint(val));
    if (old == assumed) break;
  }
}

// ---------------- conversion / transpose kernels ----------------
__global__ void cvt_a_bf16(const float* __restrict__ in,
                           unsigned short* __restrict__ out,
                           long long M, int K, int Kpad) {
  long long i = (long long)blockIdx.x * blockDim.x + threadIdx.x;
  if (i >= M * (long long)Kpad) return;
  long long m = i / Kpad;
  int k = (int)(i - m * Kpad);
  out[i] = (k < K) ? f2bf(in[m * K + k]) : (unsigned short)0;
}

__global__ void cvt_w_bf16t(const float* __restrict__ W,
                            unsigned short* __restrict__ Wt,
                            int K, int N, int Kpad, int Npad) {
  long long i = (long long)blockIdx.x * blockDim.x + threadIdx.x;
  if (i >= (long long)Npad * Kpad) return;
  int n = (int)(i / Kpad);
  int k = (int)(i % Kpad);
  Wt[i] = (n < N && k < K) ? f2bf(W[(size_t)k * N + n]) : (unsigned short)0;
}

// ---------------- WMMA GEMM ----------------
// C[M,N] = epilogue(A[M,Kpad](bf16) @ Wt[Npad,Kpad](bf16)^T + bias[N])
// MODE 0: none | 1: relu | 2: +res | 3: res + relu(.)
// JT = 16-col tiles per wave (compile-time: branch-free inner loop).
// Register double-buffered: loads for k+32 are issued before the wmmas of k.
template <int MODE, int JT>
__global__ __launch_bounds__(128)
void gemm_wmma_bf16(const unsigned short* __restrict__ A,
                    const unsigned short* __restrict__ Wt,
                    const float* __restrict__ bias,
                    const float* __restrict__ res,
                    float* __restrict__ C,
                    int M, int N, int Kpad, int res_stride)
{
  const int wave = threadIdx.x >> 5;
  const int lane = threadIdx.x & 31;
  const int half = lane >> 4;
  const int lr   = lane & 15;
  const int mtile = blockIdx.y * 4 + wave;
  if (mtile * 16 >= M) return;
  const int n0 = blockIdx.x * (JT * 16);

  v8f acc[JT];
#pragma unroll
  for (int j = 0; j < JT; ++j)
#pragma unroll
    for (int i = 0; i < 8; ++i) acc[j][i] = 0.0f;

  // A fragment (16x32 bf16): lane half h holds K = [8h..8h+7] and [16+8h..16+8h+7]
  const unsigned short* arow = A + (size_t)(mtile * 16 + lr) * Kpad + half * 8;
  // B fragment (32x16 bf16): lane half h holds K = [8h..8h+7] and [16+8h..16+8h+7] of col n
  const unsigned short* brow[JT];
#pragma unroll
  for (int j = 0; j < JT; ++j)
    brow[j] = Wt + (size_t)(n0 + j * 16 + lr) * Kpad + half * 8;

  union FragU { v16bf v; uint4 q[2]; };
  FragU fa0, fa1, fb0[JT], fb1[JT];

  auto loadFrags = [&](FragU& fa, FragU* fb, int k0) {
    fa.q[0] = *reinterpret_cast<const uint4*>(arow + k0);
    fa.q[1] = *reinterpret_cast<const uint4*>(arow + k0 + 16);
#pragma unroll
    for (int j = 0; j < JT; ++j) {
      fb[j].q[0] = *reinterpret_cast<const uint4*>(brow[j] + k0);
      fb[j].q[1] = *reinterpret_cast<const uint4*>(brow[j] + k0 + 16);
    }
  };
  auto compute = [&](FragU& fa, FragU* fb) {
#pragma unroll
    for (int j = 0; j < JT; ++j)
      acc[j] = __builtin_amdgcn_wmma_f32_16x16x32_bf16(
          false, fa.v, false, fb[j].v, (short)0, acc[j], false, false);
  };

  loadFrags(fa0, fb0, 0);                    // prologue
  int k0 = 0;
  while (true) {
    if (k0 + 32 < Kpad) loadFrags(fa1, fb1, k0 + 32);
    compute(fa0, fb0);
    k0 += 32;
    if (k0 >= Kpad) break;
    if (k0 + 32 < Kpad) loadFrags(fa0, fb0, k0 + 32);
    compute(fa1, fb1);
    k0 += 32;
    if (k0 >= Kpad) break;
  }

#pragma unroll
  for (int j = 0; j < JT; ++j) {
    const int n = n0 + j * 16 + lr;
    if (n >= N) continue;
#pragma unroll
    for (int i = 0; i < 8; ++i) {
      const int m = mtile * 16 + i + 8 * half;   // C/D layout: vgpr i, lane half
      float v = acc[j][i] + bias[n];
      if (MODE == 1)      v = v > 0.0f ? v : 0.0f;
      else if (MODE == 2) v = v + res[(size_t)m * res_stride + n];
      else if (MODE == 3) v = res[(size_t)m * res_stride + n] + (v > 0.0f ? v : 0.0f);
      C[(size_t)m * N + n] = v;
    }
  }
}

// ---------------- GATv2 edge kernels ----------------
__global__ void gat_init(float* __restrict__ amax, float* __restrict__ denom,
                         float* __restrict__ agg, int NH, long long NHC) {
  long long i = (long long)blockIdx.x * blockDim.x + threadIdx.x;
  if (i < NH) { amax[i] = -1e30f; denom[i] = 0.0f; }
  if (i < NHC) agg[i] = 0.0f;
}

__global__ void gat_alpha(const float* __restrict__ xl, const float* __restrict__ xr,
                          const float* __restrict__ att,
                          const int* __restrict__ src, const int* __restrict__ dst,
                          int H, int C,
                          float* __restrict__ alpha, float* __restrict__ amax) {
  long long i = (long long)blockIdx.x * blockDim.x + threadIdx.x;
  if (i >= (long long)D_ETOT * H) return;
  int e = (int)(i / H), h = (int)(i % H);
  int s = (e < D_E) ? src[e] : (e - D_E);
  int d = (e < D_E) ? dst[e] : (e - D_E);
  const float4* pl = reinterpret_cast<const float4*>(xl + ((size_t)s * H + h) * C);
  const float4* pr = reinterpret_cast<const float4*>(xr + ((size_t)d * H + h) * C);
  const float4* pa = reinterpret_cast<const float4*>(att + (size_t)h * C);
  float acc = 0.0f;
  for (int c = 0; c < C / 4; ++c) {
    float4 a = pl[c], b = pr[c], w = pa[c];
    float v0 = a.x + b.x; v0 = v0 > 0.f ? v0 : 0.2f * v0;
    float v1 = a.y + b.y; v1 = v1 > 0.f ? v1 : 0.2f * v1;
    float v2 = a.z + b.z; v2 = v2 > 0.f ? v2 : 0.2f * v2;
    float v3 = a.w + b.w; v3 = v3 > 0.f ? v3 : 0.2f * v3;
    acc += w.x * v0 + w.y * v1 + w.z * v2 + w.w * v3;
  }
  alpha[i] = acc;
  atomicMaxF(&amax[(size_t)d * H + h], acc);
}

__global__ void gat_expsum(float* __restrict__ alpha, const float* __restrict__ amax,
                           float* __restrict__ denom,
                           const int* __restrict__ dst, int H) {
  long long i = (long long)blockIdx.x * blockDim.x + threadIdx.x;
  if (i >= (long long)D_ETOT * H) return;
  int e = (int)(i / H), h = (int)(i % H);
  int d = (e < D_E) ? dst[e] : (e - D_E);
  float ex = __expf(alpha[i] - amax[(size_t)d * H + h]);
  alpha[i] = ex;
  atomicAdd(&denom[(size_t)d * H + h], ex);
}

// One thread per (edge, 4 consecutive features): float4 gather + 4 f32 atomics.
__global__ void gat_agg(const float* __restrict__ xl, const float* __restrict__ alpha,
                        const float* __restrict__ denom,
                        const int* __restrict__ src, const int* __restrict__ dst,
                        int H, int C, float* __restrict__ agg) {
  const int HC = H * C;
  const int HC4 = HC >> 2;
  long long i = (long long)blockIdx.x * blockDim.x + threadIdx.x;
  if (i >= (long long)D_ETOT * HC4) return;
  int e = (int)(i / HC4);
  int r = ((int)(i % HC4)) << 2;              // feature index, multiple of 4
  int h = r / C;                              // C % 4 == 0 -> r..r+3 in one head
  int s = (e < D_E) ? src[e] : (e - D_E);
  int d = (e < D_E) ? dst[e] : (e - D_E);
  float a = alpha[(size_t)e * H + h] / (denom[(size_t)d * H + h] + 1e-16f);
  float4 xv = *reinterpret_cast<const float4*>(xl + (size_t)s * HC + r);
  float* base = agg + (size_t)d * HC + r;
  atomicAdd(base + 0, xv.x * a);
  atomicAdd(base + 1, xv.y * a);
  atomicAdd(base + 2, xv.z * a);
  atomicAdd(base + 3, xv.w * a);
}

__global__ void bias_relu_k(float* __restrict__ io, const float* __restrict__ bias,
                            long long total, int HC) {
  long long i = (long long)blockIdx.x * blockDim.x + threadIdx.x;
  if (i >= total) return;
  float v = io[i] + bias[(int)(i % HC)];
  io[i] = v > 0.0f ? v : 0.0f;
}

__global__ void mean_heads_k(const float* __restrict__ agg, const float* __restrict__ bias,
                             float* __restrict__ out, int H, int C) {
  long long i = (long long)blockIdx.x * blockDim.x + threadIdx.x;
  if (i >= (long long)D_NTOT * C) return;
  int n = (int)(i / C), c = (int)(i % C);
  float s = 0.0f;
  for (int h = 0; h < H; ++h) s += agg[((size_t)n * H + h) * C + c];
  float v = s * (1.0f / H) + bias[c];
  out[i] = v > 0.0f ? v : 0.0f;
}

// ---------------- SetTransformer attention kernels ----------------
// SAB self-attention: one block per (graph, head); S=100, dh=32.
__global__ __launch_bounds__(128)
void sab_attn(const float* __restrict__ q, const float* __restrict__ k,
              const float* __restrict__ v, float* __restrict__ o)
{
  const int S = D_NPER, CT = 64, DH = 32;
  const int b = blockIdx.x >> 1;
  const int h = blockIdx.x & 1;
  __shared__ float lq[D_NPER * 32];
  __shared__ float lk[D_NPER * 32];
  __shared__ float lv[D_NPER * 32];
  for (int i = threadIdx.x; i < S * DH; i += blockDim.x) {
    int ss = i / DH, d = i % DH;
    size_t gi = ((size_t)b * S + ss) * CT + h * DH + d;
    lq[i] = q[gi]; lk[i] = k[gi]; lv[i] = v[gi];
  }
  __syncthreads();
  int qi = threadIdx.x;
  if (qi < S) {
    const float scale = 0.17677669529663687f;   // 1/sqrt(32)
    float sc[D_NPER];
    float mx = -1e30f;
    for (int kk = 0; kk < S; ++kk) {
      float sdot = 0.0f;
#pragma unroll
      for (int d = 0; d < DH; ++d) sdot += lq[qi * DH + d] * lk[kk * DH + d];
      sdot *= scale;
      sc[kk] = sdot;
      mx = fmaxf(mx, sdot);
    }
    float sum = 0.0f;
    for (int kk = 0; kk < S; ++kk) { float e_ = __expf(sc[kk] - mx); sc[kk] = e_; sum += e_; }
    float inv = 1.0f / sum;
#pragma unroll 4
    for (int d = 0; d < DH; ++d) {
      float accv = 0.0f;
      for (int kk = 0; kk < S; ++kk) accv += sc[kk] * lv[kk * DH + d];
      o[((size_t)b * S + qi) * CT + h * DH + d] = accv * inv;
    }
  }
}

// PMA attention: single query (seed) per graph, 2 heads.
__global__ __launch_bounds__(256)
void pma_attn(const float* __restrict__ qrow, const float* __restrict__ k,
              const float* __restrict__ v, float* __restrict__ o)
{
  const int S = D_NPER, CT = 64, DH = 32, H = 2;
  const int b = blockIdx.x;
  __shared__ float lk[D_NPER * 64];
  __shared__ float lv[D_NPER * 64];
  __shared__ float sc[2][D_NPER];
  __shared__ float inv[2];
  for (int i = threadIdx.x; i < S * CT; i += blockDim.x) {
    size_t gi = (size_t)b * S * CT + i;
    lk[i] = k[gi]; lv[i] = v[gi];
  }
  __syncthreads();
  const float scale = 0.17677669529663687f;
  for (int i = threadIdx.x; i < H * S; i += blockDim.x) {
    int h = i / S, kk = i % S;
    float sdot = 0.0f;
#pragma unroll
    for (int d = 0; d < DH; ++d) sdot += qrow[h * DH + d] * lk[kk * CT + h * DH + d];
    sc[h][kk] = sdot * scale;
  }
  __syncthreads();
  if (threadIdx.x < H) {
    int h = threadIdx.x;
    float mx = -1e30f;
    for (int kk = 0; kk < S; ++kk) mx = fmaxf(mx, sc[h][kk]);
    float sum = 0.0f;
    for (int kk = 0; kk < S; ++kk) { float e_ = __expf(sc[h][kk] - mx); sc[h][kk] = e_; sum += e_; }
    inv[h] = 1.0f / sum;
  }
  __syncthreads();
  if (threadIdx.x < CT) {
    int h = threadIdx.x / DH, d = threadIdx.x % DH;
    float accv = 0.0f;
    for (int kk = 0; kk < S; ++kk) accv += sc[h][kk] * lv[kk * CT + h * DH + d];
    o[(size_t)b * CT + threadIdx.x] = accv * inv[h];
  }
}

// out[n] = b[n] + sum_k x[k]*W[k*N+n]   (seed projection, tiny)
__global__ void vec_mat(const float* __restrict__ x, const float* __restrict__ W,
                        const float* __restrict__ b, float* __restrict__ out,
                        int K, int N) {
  int n = blockIdx.x * blockDim.x + threadIdx.x;
  if (n >= N) return;
  float acc = b[n];
  for (int k = 0; k < K; ++k) acc += x[k] * W[k * N + n];
  out[n] = acc;
}

// ---------------- host-side helpers ----------------
static void launch_gemm(hipStream_t s, const unsigned short* abf, const unsigned short* wt,
                        const float* bias, const float* res, float* C,
                        int M, int N, int Kpad, int mode, int res_stride) {
  const int Npad = (N + 15) & ~15;
  const int jt = (Npad % 64 == 0) ? 4 : 1;     // all shapes here: Npad%64==0 or Npad==16
  dim3 grid(Npad / (jt * 16), (M / 16 + 3) / 4);
#define GCASE(MD, JTT) gemm_wmma_bf16<MD, JTT><<<grid, 128, 0, s>>>(abf, wt, bias, res, C, M, N, Kpad, res_stride)
  if (jt == 4) {
    switch (mode) {
      case 0: GCASE(0, 4); break;
      case 1: GCASE(1, 4); break;
      case 2: GCASE(2, 4); break;
      default: GCASE(3, 4); break;
    }
  } else {
    switch (mode) {
      case 0: GCASE(0, 1); break;
      case 1: GCASE(1, 1); break;
      case 2: GCASE(2, 1); break;
      default: GCASE(3, 1); break;
    }
  }
#undef GCASE
}

static void mm(hipStream_t s, char* ws, const float* A, int M, int K,
               const float* W, const float* bias, int N,
               const float* res, int res_stride, int mode, float* C)
{
  const int Kpad = (K + 31) & ~31;
  const int Npad = (N + 15) & ~15;
  unsigned short* abf = (unsigned short*)(ws + OFF_ABF);
  unsigned short* wt  = (unsigned short*)(ws + OFF_WT);
  long long at = (long long)M * Kpad;
  cvt_a_bf16<<<(unsigned)((at + 255) / 256), 256, 0, s>>>(A, abf, M, K, Kpad);
  long long wtot = (long long)Npad * Kpad;
  cvt_w_bf16t<<<(unsigned)((wtot + 255) / 256), 256, 0, s>>>(W, wt, K, N, Kpad, Npad);
  launch_gemm(s, abf, wt, bias, res, C, M, N, Kpad, mode, res_stride);
}

static void gat_layer(hipStream_t s, char* ws,
                      const float* cur, int K, int H, int C, int concat,
                      const float* Wl, const float* bl,
                      const float* Wr, const float* br,
                      const float* att, const float* bias,
                      const int* src, const int* dst,
                      float* aggbuf, float* outbuf)
{
  const int HC = H * C;
  float* xl    = (float*)(ws + OFF_XL);
  float* xr    = (float*)(ws + OFF_XR);
  float* alpha = (float*)(ws + OFF_ALPHA);
  float* amax  = (float*)(ws + OFF_AMAX);
  float* denom = (float*)(ws + OFF_DENOM);

  mm(s, ws, cur, D_NTOT, K, Wl, bl, HC, nullptr, 0, 0, xl);
  mm(s, ws, cur, D_NTOT, K, Wr, br, HC, nullptr, 0, 0, xr);

  long long nhc = (long long)D_NTOT * HC;
  gat_init<<<(unsigned)((nhc + 255) / 256), 256, 0, s>>>(amax, denom, aggbuf,
                                                         D_NTOT * H, nhc);
  long long eh = (long long)D_ETOT * H;
  gat_alpha<<<(unsigned)((eh + 255) / 256), 256, 0, s>>>(xl, xr, att, src, dst,
                                                         H, C, alpha, amax);
  gat_expsum<<<(unsigned)((eh + 255) / 256), 256, 0, s>>>(alpha, amax, denom, dst, H);
  long long ehc4 = (long long)D_ETOT * (HC >> 2);
  gat_agg<<<(unsigned)((ehc4 + 255) / 256), 256, 0, s>>>(xl, alpha, denom, src, dst,
                                                         H, C, aggbuf);
  if (concat) {
    bias_relu_k<<<(unsigned)((nhc + 255) / 256), 256, 0, s>>>(aggbuf, bias, nhc, HC);
  } else {
    long long nc = (long long)D_NTOT * C;
    mean_heads_k<<<(unsigned)((nc + 255) / 256), 256, 0, s>>>(aggbuf, bias, outbuf, H, C);
  }
}

// ---------------- entry point ----------------
extern "C" void kernel_launch(void* const* d_in, const int* in_sizes, int n_in,
                              void* d_out, int out_size, void* d_ws, size_t ws_size,
                              hipStream_t stream) {
  (void)in_sizes; (void)n_in; (void)out_size; (void)ws_size;
  char* ws = (char*)d_ws;

  const float* x   = (const float*)d_in[0];
  const int*   ei  = (const int*)d_in[1];
  const int*   src = ei;
  const int*   dst = ei + D_E;

  auto F = [&](int i) { return (const float*)d_in[i]; };

  float* HA = (float*)(ws + OFF_HA);
  float* HB = (float*)(ws + OFF_HB);

  // ---- 4 GATv2 layers (ping-pong HA/HB, in-place bias+relu on concat layers) ----
  gat_layer(stream, ws, x,  5,   3, 128, 1, F(3),  F(4),  F(5),  F(6),  F(7),  F(8),  src, dst, HA, HA);
  gat_layer(stream, ws, HA, 384, 3, 128, 1, F(9),  F(10), F(11), F(12), F(13), F(14), src, dst, HB, HB);
  gat_layer(stream, ws, HB, 384, 3, 64,  1, F(15), F(16), F(17), F(18), F(19), F(20), src, dst, HA, HA);
  gat_layer(stream, ws, HA, 192, 3, 64,  0, F(21), F(22), F(23), F(24), F(25), F(26), src, dst, HB, HA);
  // h3 = HA  [64000, 64]

  // ---- SetTransformerAggregation ----
  float* Qb   = (float*)(ws + OFF_Q);
  float* Kb   = (float*)(ws + OFF_K);
  float* Vb   = (float*)(ws + OFF_V);
  float* AOb  = (float*)(ws + OFF_AO);
  float* ENC  = (float*)(ws + OFF_XR);
  float* TA   = (float*)(ws + OFF_TA);
  float* TB   = (float*)(ws + OFF_TB);
  float* TC   = (float*)(ws + OFF_TC);
  float* qpma = (float*)(ws + OFF_QPMA);

  // encoder SAB (heads=2) on h3
  mm(stream, ws, HA, D_NTOT, 64, F(27), F(32), 64, nullptr, 0, 0, Qb);
  mm(stream, ws, HA, D_NTOT, 64, F(28), F(33), 64, nullptr, 0, 0, Kb);
  mm(stream, ws, HA, D_NTOT, 64, F(29), F(34), 64, nullptr, 0, 0, Vb);
  sab_attn<<<D_NB * 2, 128, 0, stream>>>(Qb, Kb, Vb, AOb);
  mm(stream, ws, AOb, D_NTOT, 64, F(30), F(35), 64, HA, 64, 2, HB);   // h = x + attn@Wo+bo
  mm(stream, ws, HB,  D_NTOT, 64, F(31), F(36), 64, HB, 64, 3, ENC);  // enc = h + relu(lin(h))

  // PMA: z = relu(enc @ pma_lin + b); q from seed; k,v from z
  mm(stream, ws, ENC, D_NTOT, 64, F(37), F(38), 64, nullptr, 0, 1, HA);
  const float* seed = F(39);
  vec_mat<<<1, 64, 0, stream>>>(seed, F(40), F(45), qpma, 64, 64);    // q = seed@Wq+bq
  mm(stream, ws, HA, D_NTOT, 64, F(41), F(46), 64, nullptr, 0, 0, Kb);
  mm(stream, ws, HA, D_NTOT, 64, F(42), F(47), 64, nullptr, 0, 0, Vb);
  pma_attn<<<D_NB, 256, 0, stream>>>(qpma, Kb, Vb, TA);
  mm(stream, ws, TA, D_NB, 64, F(43), F(48), 64, seed, 0, 2, TB);     // h = seed + o@Wo+bo
  mm(stream, ws, TB, D_NB, 64, F(44), F(49), 64, TB, 64, 3, TC);      // pma_out

  // decoder SAB on [B,1,64]: softmax over one key == 1 -> attn output == v
  mm(stream, ws, TC, D_NB, 64, F(52), F(57), 64, nullptr, 0, 0, TA);  // v = d@Wv+bv
  mm(stream, ws, TA, D_NB, 64, F(53), F(58), 64, TC, 64, 2, TB);      // h = d + v@Wo+bo
  mm(stream, ws, TB, D_NB, 64, F(54), F(59), 64, TB, 64, 3, TA);      // dec_out

  // ---- MLP head ----
  mm(stream, ws, TA, D_NB, 64, F(60), F(61), 64, nullptr, 0, 0, TB);
  mm(stream, ws, TB, D_NB, 64, F(62), F(63), 64, nullptr, 0, 0, TC);
  mm(stream, ws, TC, D_NB, 64, F(64), F(65), 10, nullptr, 0, 0, (float*)d_out);
}
